// HybridBlock_50646254355299
// MI455X (gfx1250) — matrix operand order repaired and checked by reading, not compiled, not run on hardware
//
#include <hip/hip_runtime.h>

typedef __attribute__((ext_vector_type(16))) __bf16 v16bf;
typedef __attribute__((ext_vector_type(8)))  float  v8f;

#define L_SEQ   1024
#define HSZ     2048
#define PROJ_N  4256
#define DIN     2048
#define XBC     2176
#define NH      32
#define DH      64
#define DSTATE  64
#define INTER_N 8192

struct bf16x4 { __bf16 a, b, c, d; };

__device__ __forceinline__ float sigmoidf_(float v) { return 1.0f / (1.0f + __expf(-v)); }

// ---------------------------------------------------------------------------
// fp32 -> bf16 streaming convert (float4 in, 4x bf16 out)
// ---------------------------------------------------------------------------
__global__ __launch_bounds__(256)
void f32_to_bf16_kernel(const float* __restrict__ in, __bf16* __restrict__ out, int n4) {
  int i = blockIdx.x * 256 + threadIdx.x;
  if (i >= n4) return;
  float4 v = reinterpret_cast<const float4*>(in)[i];
  bf16x4 o = { (__bf16)v.x, (__bf16)v.y, (__bf16)v.z, (__bf16)v.w };
  reinterpret_cast<bf16x4*>(out)[i] = o;
}

// ---------------------------------------------------------------------------
// rmsnorm(x, w) -> bf16
// ---------------------------------------------------------------------------
__global__ __launch_bounds__(256)
void rmsnorm_bf16_kernel(const float* __restrict__ x, const float* __restrict__ w,
                         __bf16* __restrict__ out, int ncols) {
  int row = blockIdx.x;
  int tid = threadIdx.x;
  const float* xr = x + (size_t)row * ncols;
  float ss = 0.f;
  for (int c = tid; c < ncols; c += 256) { float v = xr[c]; ss += v * v; }
  __shared__ float red[256];
  red[tid] = ss; __syncthreads();
  for (int s = 128; s > 0; s >>= 1) { if (tid < s) red[tid] += red[tid + s]; __syncthreads(); }
  float rstd = rsqrtf(red[0] / (float)ncols + 1e-5f);
  __bf16* orow = out + (size_t)row * ncols;
  for (int c = tid; c < ncols; c += 256) orow[c] = (__bf16)(w[c] * xr[c] * rstd);
}

// ---------------------------------------------------------------------------
// GEMM: C[M,N] = A[M,K](bf16) * Wb[N,K](bf16)^T   (f32 accumulate via WMMA)
// 128x64 tile, 256 threads = 8 waves (4 along M x 2 along N), wave = 32x32.
// Double-buffered LDS (ping-pong), one barrier per k-step; next tile's
// global_load_b128s issued before the current tile's WMMAs.
// mode 0: store f32   mode 1: store bf16   mode 2: store resid + acc (f32)
// ---------------------------------------------------------------------------
#define LDT 40
__global__ __launch_bounds__(256)
void gemm_bf16_wmma(const __bf16* __restrict__ A, const __bf16* __restrict__ Wb,
                    float* __restrict__ outF, __bf16* __restrict__ outB,
                    const float* __restrict__ resid,
                    int M, int N, int K, int mode) {
  __shared__ __bf16 At[2][128 * LDT];   // A tile: [m][k]
  __shared__ __bf16 Bt[2][64 * LDT];    // W tile: [n][k]
  int tid    = threadIdx.x;
  int lane   = tid & 31;
  int wave   = tid >> 5;
  int waveM  = (wave >> 1) * 32;     // 0,32,64,96
  int waveN  = (wave & 1) * 32;      // 0,32
  int laneHi = lane >> 4;
  int lane15 = lane & 15;
  int tileM  = blockIdx.y * 128;
  int tileN  = blockIdx.x * 64;

  // A-tile copy mapping: 512 chunks of 8 bf16; chunk -> (row, col8)
  int ar0 = tid >> 2,          ac0 = (tid & 3) * 8;          // chunks 0..255
  int ar1 = (tid + 256) >> 2,  ac1 = ((tid + 256) & 3) * 8;  // chunks 256..511
  // W-tile copy mapping: 256 chunks of 8 bf16
  int wr = tid >> 2, wc = (tid & 3) * 8;
  bool wOK = (tileN + wr) < N;

  const __bf16* aP0 = A  + (size_t)(tileM + ar0) * K + ac0;
  const __bf16* aP1 = A  + (size_t)(tileM + ar1) * K + ac1;
  const __bf16* wP  = Wb + (size_t)(tileN + wr) * K + wc;

  // prologue: fill buffer 0
  {
    uint4 av0 = *reinterpret_cast<const uint4*>(aP0);
    uint4 av1 = *reinterpret_cast<const uint4*>(aP1);
    uint4 wv  = make_uint4(0u, 0u, 0u, 0u);
    if (wOK) wv = *reinterpret_cast<const uint4*>(wP);
    *reinterpret_cast<uint4*>(&At[0][ar0 * LDT + ac0]) = av0;
    *reinterpret_cast<uint4*>(&At[0][ar1 * LDT + ac1]) = av1;
    *reinterpret_cast<uint4*>(&Bt[0][wr * LDT + wc])   = wv;
  }
  __syncthreads();

  int nIter = K >> 5;
  v8f acc[2][2] = {};
  for (int it = 0; it < nIter; ++it) {
    int cur = it & 1, nxt = cur ^ 1;
    bool hasNext = (it + 1) < nIter;

    // issue next tile's global loads early (in flight during WMMA work)
    uint4 nav0 = make_uint4(0u, 0u, 0u, 0u);
    uint4 nav1 = make_uint4(0u, 0u, 0u, 0u);
    uint4 nwv  = make_uint4(0u, 0u, 0u, 0u);
    if (hasNext) {
      size_t ko = (size_t)(it + 1) << 5;
      nav0 = *reinterpret_cast<const uint4*>(aP0 + ko);
      nav1 = *reinterpret_cast<const uint4*>(aP1 + ko);
      if (wOK) nwv = *reinterpret_cast<const uint4*>(wP + ko);
    }

    // gather WMMA fragments per CDNA5 16-bit A/B VGPR layouts
    v16bf aFrag[2], bFrag[2];
#pragma unroll
    for (int f = 0; f < 2; ++f) {
      int arow = waveM + f * 16 + lane15;
      int bcol = waveN + f * 16 + lane15;
#pragma unroll
      for (int i = 0; i < 16; ++i) {
        int ka = (i < 8 ? i : i + 8) + (laneHi << 3);   // A: two 8-runs
        aFrag[f][i] = At[cur][arow * LDT + ka];
        int kb = (i < 8 ? i : i + 8) + (laneHi << 3);   // B: two 8-runs
        bFrag[f][i] = Bt[cur][bcol * LDT + kb];
      }
    }
#pragma unroll
    for (int fm = 0; fm < 2; ++fm)
#pragma unroll
      for (int fn = 0; fn < 2; ++fn)
        acc[fm][fn] = __builtin_amdgcn_wmma_f32_16x16x32_bf16(
            false, aFrag[fm], false, bFrag[fn], (short)0, acc[fm][fn], false, false);

    if (hasNext) {
      *reinterpret_cast<uint4*>(&At[nxt][ar0 * LDT + ac0]) = nav0;
      *reinterpret_cast<uint4*>(&At[nxt][ar1 * LDT + ac1]) = nav1;
      *reinterpret_cast<uint4*>(&Bt[nxt][wr * LDT + wc])   = nwv;
      __syncthreads();
    }
  }

  // epilogue: C/D layout -> M = vgpr + 8*laneHi, N = lane15
#pragma unroll
  for (int fm = 0; fm < 2; ++fm)
    for (int fn = 0; fn < 2; ++fn) {
      int col = tileN + waveN + fn * 16 + lane15;
      if (col >= N) continue;
      int row0 = tileM + waveM + fm * 16 + (laneHi << 3);
#pragma unroll
      for (int r = 0; r < 8; ++r) {
        float v = acc[fm][fn][r];
        size_t o = (size_t)(row0 + r) * N + col;
        if (mode == 0)      outF[o] = v;
        else if (mode == 1) outB[o] = (__bf16)v;
        else                outF[o] = resid[o] + v;
      }
    }
}

// ---------------------------------------------------------------------------
// causal depthwise conv (D_CONV=4) + bias + SiLU over the xBC slice of xz
// ---------------------------------------------------------------------------
__global__ __launch_bounds__(256)
void conv_silu_kernel(const float* __restrict__ xz, const float* __restrict__ cw,
                      const float* __restrict__ cb, float* __restrict__ xc) {
  int idx = blockIdx.x * 256 + threadIdx.x;
  if (idx >= L_SEQ * XBC) return;
  int t = idx / XBC, c = idx - t * XBC;
  float acc = cb[c];
#pragma unroll
  for (int j = 0; j < 4; ++j) {
    int tt = t - 3 + j;
    if (tt >= 0) acc += xz[(size_t)tt * PROJ_N + DIN + c] * cw[c * 4 + j];
  }
  xc[idx] = acc * sigmoidf_(acc);
}

// dt = softplus(dt_proj + dt_bias)
__global__ __launch_bounds__(256)
void dt_softplus_kernel(const float* __restrict__ xz, const float* __restrict__ dtb,
                        float* __restrict__ dt) {
  int idx = blockIdx.x * 256 + threadIdx.x;
  if (idx >= L_SEQ * NH) return;
  int t = idx >> 5, hh = idx & 31;
  float v = xz[(size_t)t * PROJ_N + (DIN + XBC) + hh] + dtb[hh];
  dt[idx] = (v > 20.f) ? v : log1pf(__expf(v));
}

// ---------------------------------------------------------------------------
// SSM scan: one workgroup per head. h[64][64] in registers (16 f32/thread).
// Time chunked by 8 to amortize barriers; y-reduce via wave32 shfl_xor.
// ---------------------------------------------------------------------------
__global__ __launch_bounds__(256)
void ssm_scan_kernel(const float* __restrict__ xc, const float* __restrict__ dt,
                     const float* __restrict__ A_log, const float* __restrict__ Dp,
                     float* __restrict__ y) {
  const int CH = 8;
  int head = blockIdx.x;
  int tid  = threadIdx.x;
  float Ah = -__expf(A_log[head]);
  float Dh = Dp[head];
  int p  = tid >> 2;          // state row this thread owns (0..63)
  int so = (tid & 3) << 4;    // state-col offset (16 cols per thread)
  float h[16];
#pragma unroll
  for (int i = 0; i < 16; ++i) h[i] = 0.f;

  __shared__ float xs[CH][DH], Bs[CH][DSTATE], Cs[CH][DSTATE], dts[CH];
  for (int t0 = 0; t0 < L_SEQ; t0 += CH) {
    for (int idx = tid; idx < CH * 192; idx += 256) {
      int st = idx / 192, e = idx - st * 192;
      const float* xrow = xc + (size_t)(t0 + st) * XBC;
      if (e < 64)       xs[st][e]       = xrow[head * DH + e];
      else if (e < 128) Bs[st][e - 64]  = xrow[DIN + e - 64];
      else              Cs[st][e - 128] = xrow[DIN + DSTATE + e - 128];
    }
    if (tid < CH) dts[tid] = dt[(t0 + tid) * NH + head];
    __syncthreads();
#pragma unroll
    for (int st = 0; st < CH; ++st) {
      float dtv = dts[st];
      float dA  = __expf(Ah * dtv);
      float xp  = xs[st][p];
      float coeff = dtv * xp;
      float part = 0.f;
#pragma unroll
      for (int i = 0; i < 16; ++i) {
        h[i] = dA * h[i] + coeff * Bs[st][so + i];
        part += h[i] * Cs[st][so + i];
      }
      part += __shfl_xor(part, 1, 32);
      part += __shfl_xor(part, 2, 32);
      if ((tid & 3) == 0)
        y[(size_t)(t0 + st) * DIN + head * DH + p] = part + Dh * xp;
    }
    __syncthreads();
  }
}

// ---------------------------------------------------------------------------
// out = rmsnorm(y * silu(z), mnorm_w) -> bf16   (z = xz[:, :DIN])
// ---------------------------------------------------------------------------
__global__ __launch_bounds__(256)
void gate_rmsnorm_kernel(const float* __restrict__ y, const float* __restrict__ xz,
                         const float* __restrict__ w, __bf16* __restrict__ out) {
  int row = blockIdx.x, tid = threadIdx.x;
  __shared__ float g[DIN];
  __shared__ float red[256];
  const float* yr = y  + (size_t)row * DIN;
  const float* zr = xz + (size_t)row * PROJ_N;
  float ss = 0.f;
  for (int c = tid; c < DIN; c += 256) {
    float z  = zr[c];
    float gv = yr[c] * (z * sigmoidf_(z));
    g[c] = gv;
    ss += gv * gv;
  }
  red[tid] = ss; __syncthreads();
  for (int s = 128; s > 0; s >>= 1) { if (tid < s) red[tid] += red[tid + s]; __syncthreads(); }
  float rstd = rsqrtf(red[0] / (float)DIN + 1e-5f);
  __bf16* o = out + (size_t)row * DIN;
  for (int c = tid; c < DIN; c += 256) o[c] = (__bf16)(w[c] * g[c] * rstd);
}

// act = silu(gate) * up  -> bf16
__global__ __launch_bounds__(256)
void silu_mul_kernel(const __bf16* __restrict__ hh, __bf16* __restrict__ act) {
  int idx = blockIdx.x * 256 + threadIdx.x;
  if (idx >= L_SEQ * INTER_N) return;
  int t = idx / INTER_N, c = idx - t * INTER_N;
  float gate = (float)hh[(size_t)t * (2 * INTER_N) + c];
  float up   = (float)hh[(size_t)t * (2 * INTER_N) + INTER_N + c];
  act[idx] = (__bf16)(gate * sigmoidf_(gate) * up);
}

// ---------------------------------------------------------------------------
extern "C" void kernel_launch(void* const* d_in, const int* in_sizes, int n_in,
                              void* d_out, int out_size, void* d_ws, size_t ws_size,
                              hipStream_t stream) {
  (void)in_sizes; (void)n_in; (void)out_size; (void)ws_size;
  const float* x          = (const float*)d_in[0];
  const float* ln1_w      = (const float*)d_in[1];
  const float* in_proj_w  = (const float*)d_in[2];
  const float* conv_w     = (const float*)d_in[3];
  const float* conv_b     = (const float*)d_in[4];
  const float* A_log      = (const float*)d_in[5];
  const float* D_param    = (const float*)d_in[6];
  const float* dt_bias    = (const float*)d_in[7];
  const float* mnorm_w    = (const float*)d_in[8];
  const float* out_proj_w = (const float*)d_in[9];
  const float* ln2_w      = (const float*)d_in[10];
  const float* mlp_in_w   = (const float*)d_in[11];
  const float* mlp_out_w  = (const float*)d_in[12];
  float* out = (float*)d_out;

  char* base = (char*)d_ws;
  size_t off = 0;
  auto alloc = [&](size_t bytes) -> void* {
    void* p = base + off;
    off = (off + bytes + 255) & ~(size_t)255;
    return p;
  };
  // activations
  __bf16* xn_bf  = (__bf16*)alloc((size_t)L_SEQ * HSZ * 2);
  float*  xz     = (float*) alloc((size_t)L_SEQ * PROJ_N * 4);
  float*  xc     = (float*) alloc((size_t)L_SEQ * XBC * 4);
  float*  dt     = (float*) alloc((size_t)L_SEQ * NH * 4);
  float*  yb     = (float*) alloc((size_t)L_SEQ * DIN * 4);
  __bf16* yg_bf  = (__bf16*)alloc((size_t)L_SEQ * DIN * 2);
  float*  x1     = (float*) alloc((size_t)L_SEQ * HSZ * 4);
  __bf16* xn2_bf = (__bf16*)alloc((size_t)L_SEQ * HSZ * 2);
  __bf16* hh_bf  = (__bf16*)alloc((size_t)L_SEQ * 2 * INTER_N * 2);
  __bf16* act_bf = (__bf16*)alloc((size_t)L_SEQ * INTER_N * 2);
  // bf16 weights (converted once per launch)
  __bf16* w1_bf  = (__bf16*)alloc((size_t)PROJ_N * HSZ * 2);
  __bf16* w2_bf  = (__bf16*)alloc((size_t)HSZ * DIN * 2);
  __bf16* w3_bf  = (__bf16*)alloc((size_t)(2 * INTER_N) * HSZ * 2);
  __bf16* w4_bf  = (__bf16*)alloc((size_t)HSZ * INTER_N * 2);

  // 0) weight fp32 -> bf16 (streaming, float4 granularity)
  auto cvt = [&](const float* src, __bf16* dst, size_t n) {
    int n4 = (int)(n / 4);
    f32_to_bf16_kernel<<<(n4 + 255) / 256, 256, 0, stream>>>(src, dst, n4);
  };
  cvt(in_proj_w,  w1_bf, (size_t)PROJ_N * HSZ);
  cvt(out_proj_w, w2_bf, (size_t)HSZ * DIN);
  cvt(mlp_in_w,   w3_bf, (size_t)(2 * INTER_N) * HSZ);
  cvt(mlp_out_w,  w4_bf, (size_t)HSZ * INTER_N);

  // 1) xn = rmsnorm(x, ln1_w) -> bf16
  rmsnorm_bf16_kernel<<<L_SEQ, 256, 0, stream>>>(x, ln1_w, xn_bf, HSZ);
  // 2) xz = xn @ in_proj_w^T  (f32 out)
  gemm_bf16_wmma<<<dim3((PROJ_N + 63) / 64, L_SEQ / 128), 256, 0, stream>>>(
      xn_bf, w1_bf, xz, nullptr, nullptr, L_SEQ, PROJ_N, HSZ, 0);
  // 3) conv + silu on xBC slice; 4) dt softplus
  conv_silu_kernel<<<(L_SEQ * XBC + 255) / 256, 256, 0, stream>>>(xz, conv_w, conv_b, xc);
  dt_softplus_kernel<<<(L_SEQ * NH + 255) / 256, 256, 0, stream>>>(xz, dt_bias, dt);
  // 5) sequential SSM scan (one WG per head)
  ssm_scan_kernel<<<NH, 256, 0, stream>>>(xc, dt, A_log, D_param, yb);
  // 6) yg = rmsnorm(y * silu(z), mnorm_w) -> bf16
  gate_rmsnorm_kernel<<<L_SEQ, 256, 0, stream>>>(yb, xz, mnorm_w, yg_bf);
  // 7) x1 = x + yg @ out_proj_w^T  (fused residual)
  gemm_bf16_wmma<<<dim3(HSZ / 64, L_SEQ / 128), 256, 0, stream>>>(
      yg_bf, w2_bf, x1, nullptr, x, L_SEQ, HSZ, DIN, 2);
  // 8) xn2 = rmsnorm(x1, ln2_w) -> bf16
  rmsnorm_bf16_kernel<<<L_SEQ, 256, 0, stream>>>(x1, ln2_w, xn2_bf, HSZ);
  // 9) hh = xn2 @ mlp_in_w^T -> bf16 (gate|up)
  gemm_bf16_wmma<<<dim3((2 * INTER_N) / 64, L_SEQ / 128), 256, 0, stream>>>(
      xn2_bf, w3_bf, nullptr, hh_bf, nullptr, L_SEQ, 2 * INTER_N, HSZ, 1);
  // 10) act = silu(gate) * up -> bf16
  silu_mul_kernel<<<(L_SEQ * INTER_N + 255) / 256, 256, 0, stream>>>(hh_bf, act_bf);
  // 11) out = x1 + act @ mlp_out_w^T  (fused residual, final output)
  gemm_bf16_wmma<<<dim3(HSZ / 64, L_SEQ / 128), 256, 0, stream>>>(
      act_bf, w4_bf, out, nullptr, x1, L_SEQ, HSZ, INTER_N, 2);
}
